// DeformConv3d_33930241638898
// MI455X (gfx1250) — matrix operand order repaired and checked by reading, not compiled, not run on hardware
//
#include <hip/hip_runtime.h>
#include <math.h>

typedef _Float16 v16h __attribute__((ext_vector_type(16)));
typedef _Float16 v8h  __attribute__((ext_vector_type(8)));
typedef float    v8f  __attribute__((ext_vector_type(8)));

#define CIN_   32
#define COUT_  64
#define KTAPS  27
#define D_     8
#define H_     48
#define W_     48
#define SPAT   (D_ * H_ * W_)      /* 18432 */
#define BATCH  2
#define NPOS   (BATCH * SPAT)      /* 36864 */
#define WAVES_PER_BLOCK 8

__device__ __forceinline__ int iclamp(int v, int lo, int hi) {
    return v < lo ? lo : (v > hi ? hi : v);
}

__global__ __launch_bounds__(256)
void deform_conv3d_wmma_kernel(const float* __restrict__ x,
                               const float* __restrict__ off,
                               const float* __restrict__ msk,
                               const float* __restrict__ wgt,
                               const float* __restrict__ bias,
                               float* __restrict__ out)
{
    extern __shared__ _Float16 smem[];
    _Float16* Wlds = smem;                               // [KTAPS][COUT_][CIN_] = 55296 halfs
    const int tid  = threadIdx.x;
    const int lane = tid & 31;
    const int wid  = tid >> 5;
    _Float16* As = smem + KTAPS * COUT_ * CIN_ + wid * (16 * 32);  // per-wave 16x32 A tile

    // ---- Stage weights (COUT, CIN, 27) f32 -> LDS f16 as W[tap][cout][cin] ----
    for (int i = tid; i < COUT_ * CIN_ * KTAPS; i += 256) {
        int o   = i / (CIN_ * KTAPS);
        int rem = i - o * (CIN_ * KTAPS);
        int c   = rem / KTAPS;
        int k   = rem - c * KTAPS;
        Wlds[(k * COUT_ + o) * CIN_ + c] = (_Float16)wgt[i];
    }
    __syncthreads();

    const int tile = blockIdx.x * WAVES_PER_BLOCK + wid;   // 0..2303
    const int base = tile * 16;                            // flat output position base

    // ---- per-lane position params: 2 lanes per position, 16 channels each ----
    const int p      = lane >> 1;           // position within tile: 0..15
    const int chalf  = (lane & 1) * 16;     // channel half: 0 or 16
    const int pos    = base + p;
    const int b      = pos / SPAT;
    const int sp     = pos - b * SPAT;
    const int d      = sp / (H_ * W_);
    const int r      = sp - d * (H_ * W_);
    const int h      = r / W_;
    const int w      = r - h * W_;

    const float* offB = off + (size_t)b * (2 * KTAPS) * SPAT + sp;
    const float* mskB = msk + (size_t)b * KTAPS * SPAT + sp;
    const size_t xb0  = (size_t)b * CIN_ * SPAT;

    v8f acc[4] = {v8f{}, v8f{}, v8f{}, v8f{}};

    const int row  = lane & 15;                 // A row / B-D column
    const int kbA  = (lane & 16) ? 8 : 0;       // A K-base quirk per ISA layout
    const int kbB  = (lane & 16) ? 8 : 0;       // B K-base (same table order as A)

    for (int tap = 0; tap < KTAPS; ++tap) {
        const int kd = tap / 9;
        const int kh = (tap / 3) % 3;
        const int kw = tap % 3;

        const float oh = offB[(size_t)(2 * tap) * SPAT];
        const float ow = offB[(size_t)(2 * tap + 1) * SPAT];
        const float m  = mskB[(size_t)tap * SPAT];

        const int  dpos = d + kd - 1;
        const bool vd   = (dpos >= 0) & (dpos < D_);
        const int  didx = iclamp(dpos, 0, D_ - 1);

        const float hs  = (float)(h + kh - 1) + oh;
        const float ws  = (float)(w + kw - 1) + ow;
        const float h0f = floorf(hs), w0f = floorf(ws);
        const float lh  = hs - h0f,   lw  = ws - w0f;
        const int h0 = (int)h0f, w0 = (int)w0f;
        const int h1 = h0 + 1,   w1 = w0 + 1;
        const bool ih0 = (h0 >= 0) & (h0 < H_), ih1 = (h1 >= 0) & (h1 < H_);
        const bool iw0 = (w0 >= 0) & (w0 < W_), iw1 = (w1 >= 0) & (w1 < W_);
        const float w00 = (1.f - lh) * (1.f - lw) * m * (float)(ih0 & iw0 & vd);
        const float w01 = (1.f - lh) * lw         * m * (float)(ih0 & iw1 & vd);
        const float w10 = lh * (1.f - lw)         * m * (float)(ih1 & iw0 & vd);
        const float w11 = lh * lw                 * m * (float)(ih1 & iw1 & vd);
        const int hc0 = iclamp(h0, 0, H_ - 1), hc1 = iclamp(h1, 0, H_ - 1);
        const int wc0 = iclamp(w0, 0, W_ - 1), wc1 = iclamp(w1, 0, W_ - 1);
        const int c00 = hc0 * W_ + wc0, c01 = hc0 * W_ + wc1;
        const int c10 = hc1 * W_ + wc0, c11 = hc1 * W_ + wc1;

        const float* xb = x + xb0 + (size_t)didx * (H_ * W_) + (size_t)chalf * SPAT;

        // ---- gather + bilinear interp for 16 channels ----
        v8h s0, s1;
        #pragma unroll
        for (int j = 0; j < 16; ++j) {
            const float* xc = xb + (size_t)j * SPAT;
            const float v = w00 * xc[c00] + w01 * xc[c01] + w10 * xc[c10] + w11 * xc[c11];
            if (j < 8) s0[j]     = (_Float16)v;
            else       s1[j - 8] = (_Float16)v;
        }
        *(v8h*)(As + p * 32 + chalf)     = s0;   // ds_store_b128
        *(v8h*)(As + p * 32 + chalf + 8) = s1;   // ds_store_b128
        asm volatile("s_wait_dscnt 0" ::: "memory");

        // ---- A fragment: 16-bit A 16x32 layout (ISA 7.12.2) ----
        v8h a0 = *(const v8h*)(As + row * 32 + kbA);
        v8h a1 = *(const v8h*)(As + row * 32 + kbA + 16);
        v16h a;
        #pragma unroll
        for (int e = 0; e < 8; ++e) { a[e] = a0[e]; a[8 + e] = a1[e]; }

        // ---- 4 N-tiles of B from LDS weights, WMMA accumulate ----
        #pragma unroll
        for (int t = 0; t < 4; ++t) {
            const _Float16* bp = Wlds + ((size_t)tap * COUT_ + t * 16 + row) * CIN_ + kbB;
            v8h b0 = *(const v8h*)(bp);
            v8h b1 = *(const v8h*)(bp + 16);
            v16h bb;
            #pragma unroll
            for (int e = 0; e < 8; ++e) { bb[e] = b0[e]; bb[8 + e] = b1[e]; }
            acc[t] = __builtin_amdgcn_wmma_f32_16x16x32_f16(
                         false, a, false, bb, (short)0, acc[t], false, false);
        }
    }

    // ---- epilogue: bias + store (rows are consecutive w positions) ----
    const int hi8 = (lane & 16) ? 8 : 0;
    const int ob  = base / SPAT;
    const int osp = base - ob * SPAT;
    #pragma unroll
    for (int t = 0; t < 4; ++t) {
        const int cout = t * 16 + row;
        const float bv = bias[cout];
        float* op = out + (size_t)ob * COUT_ * SPAT + (size_t)cout * SPAT + osp + hi8;
        float4 lo = make_float4(acc[t][0] + bv, acc[t][1] + bv, acc[t][2] + bv, acc[t][3] + bv);
        float4 hi = make_float4(acc[t][4] + bv, acc[t][5] + bv, acc[t][6] + bv, acc[t][7] + bv);
        *(float4*)(op)     = lo;
        *(float4*)(op + 4) = hi;
    }
}

extern "C" void kernel_launch(void* const* d_in, const int* in_sizes, int n_in,
                              void* d_out, int out_size, void* d_ws, size_t ws_size,
                              hipStream_t stream) {
    (void)in_sizes; (void)n_in; (void)out_size; (void)d_ws; (void)ws_size;
    const float* x    = (const float*)d_in[0];
    const float* off  = (const float*)d_in[1];
    const float* msk  = (const float*)d_in[2];
    const float* wgt  = (const float*)d_in[3];
    const float* bias = (const float*)d_in[4];
    float* out = (float*)d_out;

    const size_t shmem = (size_t)(KTAPS * COUT_ * CIN_ + WAVES_PER_BLOCK * 16 * 32) * 2; // 118784 B
    const int nblocks = NPOS / 16 / WAVES_PER_BLOCK;  // 288
    deform_conv3d_wmma_kernel<<<nblocks, 256, shmem, stream>>>(x, off, msk, wgt, bias, out);
}
